// RowSelfAttention_87771951661325
// MI455X (gfx1250) — matrix operand order repaired and checked by reading, not compiled, not run on hardware
//
#include <hip/hip_runtime.h>
#include <hip/hip_bf16.h>
#include <math.h>
#include <stdint.h>

// ---------------------------------------------------------------------------
// Tied row self-attention (MSA row attention), bf16 WMMA pipeline for gfx1250
// Double-buffered async global->LDS staging (ASYNCcnt) + plain K-order LDS
// fragment layouts so every WMMA operand is two ds_load_b128, zero repacking.
// ---------------------------------------------------------------------------

typedef __bf16 bf16;
typedef __attribute__((ext_vector_type(16))) __bf16 v16bf;
typedef __attribute__((ext_vector_type(8)))  __bf16 v8bf;
typedef __attribute__((ext_vector_type(8)))  float  v8f;

#define R_DIM 128
#define C_DIM 512
#define E_DIM 768
#define H_DIM 12
#define D_DIM 64
#define M_TOT (R_DIM * C_DIM)   // 65536 tokens
#define N_QKV (3 * E_DIM)       // 2304 fused qkv columns
#define KR    (R_DIM * D_DIM)   // 8192  (r,d) contraction for attention

// GEMM tiling: 128x128 block tile, K-step 32 (one bf16 WMMA K), 8 waves (4x2),
// each wave: 32x64 -> 2x4 grid of 16x16 WMMA tiles. Double-buffered LDS.
#define BM 128
#define BN 128
#define BK 32
#define AS_LD 40    // 32 + 8 pad; 80B row stride keeps 16B chunk alignment

// scaling = head_dim^-0.5 / sqrt(num_rows) = 0.125 / sqrt(128); folded into
// the q columns of Wcat/bcat at conversion time.
#define Q_SCALE 0.011048543456039806f

#define USE_ASYNC 1

// ----------------------------- async helpers -------------------------------

#if USE_ASYNC
__device__ __forceinline__ void async_copy_b128(unsigned lds_addr, const bf16* g) {
  asm volatile("global_load_async_to_lds_b128 %0, %1, off"
               :: "v"(lds_addr), "v"((unsigned long long)(uintptr_t)g)
               : "memory");
}
__device__ __forceinline__ void async_wait0() {
  asm volatile("s_wait_asynccnt 0x0" ::: "memory");
}
#else
__device__ __forceinline__ void async_wait0() {}
#endif

// ----------------------------- fragment loads ------------------------------
// Every fragment is 32 contiguous bytes in LDS (16B aligned) -> two
// ds_load_b128 straight into the WMMA operand register pair.

__device__ __forceinline__ v16bf load_frag32(const bf16* p) {
  v16bf r;
  __builtin_memcpy(&r, p, 32);
  return r;
}

// A tile rows in plain K order: lane half picks K 0-15 / K 16-31.
__device__ __forceinline__ v16bf load_a_frag(const bf16* As, int mo, int lane) {
  int row = mo + (lane & 15);
  return load_frag32(As + row * AS_LD + (lane >> 4) * 16);
}

// B tile stored [N][K] plain order: lane half picks K 0-15 / K 16-31
__device__ __forceinline__ v16bf load_b_frag(const bf16* Bs2, int no, int lane) {
  int col = no + (lane & 15);
  return load_frag32(Bs2 + col * AS_LD + (lane >> 4) * 16);
}

// ------------------------------ LDS staging --------------------------------

// A tile: 128 rows x 32 K (contiguous in memory), plain placement.
__device__ __forceinline__ void stage_a_bf16(bf16* As, const bf16* g, long ld, int t) {
  int row = t >> 1, s = t & 1;                 // s=0: K0-15, s=1: K16-31
  const bf16* src = g + (long)row * ld + s * 16;
  int o0 = s ? 16 : 0, o1 = s ? 24 : 8;        // plain chunk offsets
#if USE_ASYNC
  unsigned base = (unsigned)(uintptr_t)(As + row * AS_LD);
  async_copy_b128(base + o0 * 2, src);
  async_copy_b128(base + o1 * 2, src + 8);
#else
  v8bf a0 = *(const v8bf*)src;
  v8bf a1 = *(const v8bf*)(src + 8);
  *(v8bf*)(As + row * AS_LD + o0) = a0;
  *(v8bf*)(As + row * AS_LD + o1) = a1;
#endif
}

// A tile from fp32 source, converted to bf16, plain placement.
__device__ __forceinline__ void stage_a_f32(bf16* As, const float* g, long ld, int t) {
  int row = t >> 1, s = t & 1;
  const float4* src = (const float4*)(g + (long)row * ld + s * 16);
  float4 f0 = src[0], f1 = src[1], f2 = src[2], f3 = src[3];
  v8bf o0, o1;
  o0[0] = (bf16)f0.x; o0[1] = (bf16)f0.y; o0[2] = (bf16)f0.z; o0[3] = (bf16)f0.w;
  o0[4] = (bf16)f1.x; o0[5] = (bf16)f1.y; o0[6] = (bf16)f1.z; o0[7] = (bf16)f1.w;
  o1[0] = (bf16)f2.x; o1[1] = (bf16)f2.y; o1[2] = (bf16)f2.z; o1[3] = (bf16)f2.w;
  o1[4] = (bf16)f3.x; o1[5] = (bf16)f3.y; o1[6] = (bf16)f3.z; o1[7] = (bf16)f3.w;
  *(v8bf*)(As + row * AS_LD + (s ? 16 : 0))  = o0;
  *(v8bf*)(As + row * AS_LD + (s ? 24 : 8)) = o1;
}

// B tile [N][K] from a row-major-in-K source (scores path): plain placement.
__device__ __forceinline__ void stage_b_rows(bf16* Bs2, const bf16* g, long ld, int t) {
  int row = t >> 1, s = t & 1;                 // row = n column index
  const bf16* src = g + (long)row * ld + s * 16;
#if USE_ASYNC
  unsigned base = (unsigned)(uintptr_t)(Bs2 + row * AS_LD + s * 16);
  async_copy_b128(base, src);
  async_copy_b128(base + 16, src + 8);
#else
  v8bf a0 = *(const v8bf*)src;
  v8bf a1 = *(const v8bf*)(src + 8);
  *(v8bf*)(Bs2 + row * AS_LD + s * 16)     = a0;
  *(v8bf*)(Bs2 + row * AS_LD + s * 16 + 8) = a1;
#endif
}

// B tile [N][K] from a [K][N] source: wave-coalesced strided u16 gathers.
__device__ __forceinline__ void stage_b_transpose(bf16* Bs2, const bf16* colptr,
                                                  long ld, int n, int kh) {
  v8bf o0, o1;
#pragma unroll
  for (int e = 0; e < 8; ++e) o0[e] = colptr[(long)e * ld];
#pragma unroll
  for (int e = 0; e < 8; ++e) o1[e] = colptr[(long)(e + 8) * ld];
  __builtin_prefetch(colptr + (long)BK * ld, 0, 1);
  *(v8bf*)(Bs2 + n * AS_LD + kh)     = o0;
  *(v8bf*)(Bs2 + n * AS_LD + kh + 8) = o1;
}

// ------------------------- weight conversion kernel ------------------------

__global__ __launch_bounds__(256)
void convert_weights(const float* __restrict__ Wq, const float* __restrict__ Wk,
                     const float* __restrict__ Wv, const float* __restrict__ Wo,
                     const float* __restrict__ bq, const float* __restrict__ bk,
                     const float* __restrict__ bv,
                     bf16* __restrict__ Wcat, bf16* __restrict__ WoB,
                     float* __restrict__ bcat) {
  long idx = (long)blockIdx.x * 256 + threadIdx.x;
  const long NW = (long)E_DIM * N_QKV;
  if (idx < NW) {
    long k = idx / N_QKV, n = idx % N_QKV;
    float v;
    if (n < E_DIM)            v = Wq[k * E_DIM + n] * Q_SCALE;   // fold scale
    else if (n < 2 * E_DIM)   v = Wk[k * E_DIM + (n - E_DIM)];
    else                      v = Wv[k * E_DIM + (n - 2 * E_DIM)];
    Wcat[idx] = (bf16)v;
  } else if (idx < NW + (long)E_DIM * E_DIM) {
    long j = idx - NW;
    WoB[j] = (bf16)Wo[j];
  }
  if (idx < N_QKV) {
    float b = (idx < E_DIM) ? bq[idx] * Q_SCALE
            : (idx < 2 * E_DIM ? bk[idx - E_DIM] : bv[idx - 2 * E_DIM]);
    bcat[idx] = b;
  }
}

// --------------------------- WMMA micro-kernel -----------------------------

__device__ __forceinline__ void wmma_step(const bf16* As, const bf16* Bs2,
                                          int wm, int wn, int lane,
                                          v8f acc[2][4]) {
  v16bf a[2], b[4];
#pragma unroll
  for (int mt = 0; mt < 2; ++mt) a[mt] = load_a_frag(As, wm * 32 + mt * 16, lane);
#pragma unroll
  for (int nt = 0; nt < 4; ++nt) b[nt] = load_b_frag(Bs2, wn * 64 + nt * 16, lane);
#pragma unroll
  for (int mt = 0; mt < 2; ++mt)
#pragma unroll
    for (int nt = 0; nt < 4; ++nt)
      acc[mt][nt] = __builtin_amdgcn_wmma_f32_16x16x32_bf16(
          false, a[mt], false, b[nt], (short)0, acc[mt][nt], false, false);
}

// ------------------------------ fused QKV GEMM -----------------------------
// qkv[m][n] = sum_k x[m][k] * Wcat[k][n] + bcat[n]  (bf16 out, q pre-scaled)

__global__ __launch_bounds__(256)
void qkv_gemm(const float* __restrict__ x, const bf16* __restrict__ Wcat,
              const float* __restrict__ bcat, bf16* __restrict__ qkv) {
  __shared__ bf16 As[2][BM * AS_LD];
  __shared__ bf16 Bs2[2][BN * AS_LD];
  int t = threadIdx.x, lane = t & 31, w = t >> 5;
  int wm = w >> 1, wn = w & 1;
  int m0 = blockIdx.y * BM, n0 = blockIdx.x * BN;
  int bn = t & 127, bkh = (t >> 7) * 16;       // transpose-staging coords
  const int T = E_DIM / BK;

  v8f acc[2][4] = {};
  stage_a_f32(As[0], x + (long)m0 * E_DIM, E_DIM, t);
  stage_b_transpose(Bs2[0], Wcat + (long)bkh * N_QKV + n0 + bn, N_QKV, bn, bkh);
  for (int i = 0; i < T; ++i) {
    int cur = i & 1;
    async_wait0();
    __syncthreads();
    if (i + 1 < T) {
      int k0 = (i + 1) * BK;
      stage_a_f32(As[cur ^ 1], x + (long)m0 * E_DIM + k0, E_DIM, t);
      stage_b_transpose(Bs2[cur ^ 1], Wcat + (long)(k0 + bkh) * N_QKV + n0 + bn,
                        N_QKV, bn, bkh);
    }
    wmma_step(As[cur], Bs2[cur], wm, wn, lane, acc);
  }

#pragma unroll
  for (int mt = 0; mt < 2; ++mt)
#pragma unroll
    for (int nt = 0; nt < 4; ++nt)
#pragma unroll
      for (int e = 0; e < 8; ++e) {
        int gm = m0 + wm * 32 + mt * 16 + (lane >> 4) * 8 + e;
        int gn = n0 + wn * 64 + nt * 16 + (lane & 15);
        qkv[(long)gm * N_QKV + gn] = (bf16)(acc[mt][nt][e] + bcat[gn]);
      }
}

// ------------------------------ scores GEMM --------------------------------
// S[h][i][j] = sum_{r,d} q[r,i,h,d] * k[r,j,h,d]   (K-dim = 8192), fp32 out

__global__ __launch_bounds__(256)
void scores_gemm(const bf16* __restrict__ qkv, float* __restrict__ scores) {
  __shared__ bf16 As[2][BM * AS_LD];
  __shared__ bf16 Bs2[2][BN * AS_LD];
  int t = threadIdx.x, lane = t & 31, w = t >> 5;
  int wm = w >> 1, wn = w & 1;
  int h = blockIdx.z;
  int i0 = blockIdx.y * BM, j0 = blockIdx.x * BN;
  const int T = KR / BK;
  const bf16* qbase = qkv + (long)i0 * N_QKV + h * D_DIM;
  const bf16* kbase = qkv + (long)j0 * N_QKV + E_DIM + h * D_DIM;

  v8f acc[2][4] = {};
  stage_a_bf16(As[0], qbase, N_QKV, t);
  stage_b_rows(Bs2[0], kbase, N_QKV, t);
  for (int i = 0; i < T; ++i) {
    int cur = i & 1;
    async_wait0();
    __syncthreads();
    if (i + 1 < T) {
      int kt = (i + 1) * BK;
      int r = kt >> 6, db = kt & 63;   // BK=32 stays inside one row r (D=64)
      long roff = (long)r * C_DIM * N_QKV + db;
      stage_a_bf16(As[cur ^ 1], qbase + roff, N_QKV, t);
      stage_b_rows(Bs2[cur ^ 1], kbase + roff, N_QKV, t);
    }
    wmma_step(As[cur], Bs2[cur], wm, wn, lane, acc);
  }

#pragma unroll
  for (int mt = 0; mt < 2; ++mt)
#pragma unroll
    for (int nt = 0; nt < 4; ++nt)
#pragma unroll
      for (int e = 0; e < 8; ++e) {
        int gi = i0 + wm * 32 + mt * 16 + (lane >> 4) * 8 + e;
        int gj = j0 + wn * 64 + nt * 16 + (lane & 15);
        scores[((long)h * C_DIM + gi) * C_DIM + gj] = acc[mt][nt][e];
      }
}

// ------------------------------- softmax -----------------------------------

__global__ __launch_bounds__(256)
void softmax_kernel(const float* __restrict__ s, bf16* __restrict__ p) {
  long row = blockIdx.x;
  const float* sr = s + row * C_DIM;
  int t = threadIdx.x;
  float v0 = sr[t], v1 = sr[t + 256];
  __shared__ float red[8];

  float m = fmaxf(v0, v1);
#pragma unroll
  for (int off = 16; off > 0; off >>= 1) m = fmaxf(m, __shfl_xor(m, off));
  if ((t & 31) == 0) red[t >> 5] = m;
  __syncthreads();
  float bm = red[0];
#pragma unroll
  for (int i = 1; i < 8; ++i) bm = fmaxf(bm, red[i]);

  float e0 = __expf(v0 - bm), e1 = __expf(v1 - bm);
  float ss = e0 + e1;
#pragma unroll
  for (int off = 16; off > 0; off >>= 1) ss += __shfl_xor(ss, off);
  __syncthreads();
  if ((t & 31) == 0) red[t >> 5] = ss;
  __syncthreads();
  float tot = 0.f;
#pragma unroll
  for (int i = 0; i < 8; ++i) tot += red[i];
  float inv = 1.0f / tot;
  p[row * C_DIM + t]       = (bf16)(e0 * inv);
  p[row * C_DIM + t + 256] = (bf16)(e1 * inv);
}

// ------------------------------ context GEMM -------------------------------
// ctx[r,i,h,d] = sum_j P[h][i][j] * v[r,j,h,d]; per head: [512 x 8192] output

__global__ __launch_bounds__(256)
void ctx_gemm(const bf16* __restrict__ probs, const bf16* __restrict__ qkv,
              bf16* __restrict__ ctx) {
  __shared__ bf16 As[2][BM * AS_LD];
  __shared__ bf16 Bs2[2][BN * AS_LD];
  int t = threadIdx.x, lane = t & 31, w = t >> 5;
  int wm = w >> 1, wn = w & 1;
  int h = blockIdx.z;
  int i0 = blockIdx.y * BM, n0 = blockIdx.x * BN;
  int rr = n0 >> 6;                     // N tile covers rows rr, rr+1 (D=64)
  int bn = t & 127, bkh = (t >> 7) * 16;
  int br = rr + (bn >> 6), bd = bn & 63;
  const int T = C_DIM / BK;
  const bf16* abase = probs + ((long)h * C_DIM + i0) * C_DIM;
  const bf16* vbase = qkv + ((long)br * C_DIM + bkh) * N_QKV
                      + 2 * E_DIM + h * D_DIM + bd;

  v8f acc[2][4] = {};
  stage_a_bf16(As[0], abase, C_DIM, t);
  stage_b_transpose(Bs2[0], vbase, N_QKV, bn, bkh);
  for (int i = 0; i < T; ++i) {
    int cur = i & 1;
    async_wait0();
    __syncthreads();
    if (i + 1 < T) {
      int j0 = (i + 1) * BK;
      stage_a_bf16(As[cur ^ 1], abase + j0, C_DIM, t);
      stage_b_transpose(Bs2[cur ^ 1], vbase + (long)j0 * N_QKV, N_QKV, bn, bkh);
    }
    wmma_step(As[cur], Bs2[cur], wm, wn, lane, acc);
  }

#pragma unroll
  for (int mt = 0; mt < 2; ++mt)
#pragma unroll
    for (int nt = 0; nt < 4; ++nt)
#pragma unroll
      for (int e = 0; e < 8; ++e) {
        int gi = i0 + wm * 32 + mt * 16 + (lane >> 4) * 8 + e;
        int ln = wn * 64 + nt * 16 + (lane & 15);
        int r  = rr + (ln >> 6), d = ln & 63;
        ctx[((long)(r * C_DIM) + gi) * E_DIM + h * D_DIM + d] = (bf16)acc[mt][nt][e];
      }
}

// --------------------------- output projection -----------------------------
// out[m][e] = sum_k ctx[m][k] * Wo[k][e] + bo[e], fp32 out

__global__ __launch_bounds__(256)
void out_gemm(const bf16* __restrict__ ctx, const bf16* __restrict__ WoB,
              const float* __restrict__ bo, float* __restrict__ out) {
  __shared__ bf16 As[2][BM * AS_LD];
  __shared__ bf16 Bs2[2][BN * AS_LD];
  int t = threadIdx.x, lane = t & 31, w = t >> 5;
  int wm = w >> 1, wn = w & 1;
  int m0 = blockIdx.y * BM, n0 = blockIdx.x * BN;
  int bn = t & 127, bkh = (t >> 7) * 16;
  const int T = E_DIM / BK;

  v8f acc[2][4] = {};
  stage_a_bf16(As[0], ctx + (long)m0 * E_DIM, E_DIM, t);
  stage_b_transpose(Bs2[0], WoB + (long)bkh * E_DIM + n0 + bn, E_DIM, bn, bkh);
  for (int i = 0; i < T; ++i) {
    int cur = i & 1;
    async_wait0();
    __syncthreads();
    if (i + 1 < T) {
      int k0 = (i + 1) * BK;
      stage_a_bf16(As[cur ^ 1], ctx + (long)m0 * E_DIM + k0, E_DIM, t);
      stage_b_transpose(Bs2[cur ^ 1], WoB + (long)(k0 + bkh) * E_DIM + n0 + bn,
                        E_DIM, bn, bkh);
    }
    wmma_step(As[cur], Bs2[cur], wm, wn, lane, acc);
  }

#pragma unroll
  for (int mt = 0; mt < 2; ++mt)
#pragma unroll
    for (int nt = 0; nt < 4; ++nt)
#pragma unroll
      for (int e = 0; e < 8; ++e) {
        int gm = m0 + wm * 32 + mt * 16 + (lane >> 4) * 8 + e;
        int gn = n0 + wn * 64 + nt * 16 + (lane & 15);
        out[(long)gm * E_DIM + gn] = acc[mt][nt][e] + bo[gn];
      }
}

// ------------------------------- launcher ----------------------------------

extern "C" void kernel_launch(void* const* d_in, const int* in_sizes, int n_in,
                              void* d_out, int out_size, void* d_ws, size_t ws_size,
                              hipStream_t stream) {
  const float* x  = (const float*)d_in[0];
  const float* Wq = (const float*)d_in[1];
  const float* bq = (const float*)d_in[2];
  const float* Wk = (const float*)d_in[3];
  const float* bk = (const float*)d_in[4];
  const float* Wv = (const float*)d_in[5];
  const float* bv = (const float*)d_in[6];
  const float* Wo = (const float*)d_in[7];
  const float* bo = (const float*)d_in[8];
  float* out = (float*)d_out;

  // workspace carve-up (~406 MiB total)
  char* ws = (char*)d_ws;
  bf16* qkv = (bf16*)ws;                 ws += (size_t)M_TOT * N_QKV * sizeof(bf16);
  bf16* Wcat = (bf16*)ws;                ws += (size_t)E_DIM * N_QKV * sizeof(bf16);
  bf16* WoB = (bf16*)ws;                 ws += (size_t)E_DIM * E_DIM * sizeof(bf16);
  float* bcat = (float*)ws;              ws += (size_t)N_QKV * sizeof(float);
  float* scores = (float*)ws;            ws += (size_t)H_DIM * C_DIM * C_DIM * sizeof(float);
  bf16* probs = (bf16*)ws;               ws += (size_t)H_DIM * C_DIM * C_DIM * sizeof(bf16);
  bf16* ctx = (bf16*)ws;

  {
    long tot = (long)E_DIM * N_QKV + (long)E_DIM * E_DIM;
    convert_weights<<<(unsigned)((tot + 255) / 256), 256, 0, stream>>>(
        Wq, Wk, Wv, Wo, bq, bk, bv, Wcat, WoB, bcat);
  }
  qkv_gemm<<<dim3(N_QKV / BN, M_TOT / BM), 256, 0, stream>>>(x, Wcat, bcat, qkv);
  scores_gemm<<<dim3(C_DIM / BN, C_DIM / BM, H_DIM), 256, 0, stream>>>(qkv, scores);
  softmax_kernel<<<H_DIM * C_DIM, 256, 0, stream>>>(scores, probs);
  ctx_gemm<<<dim3(KR / BN, C_DIM / BM, H_DIM), 256, 0, stream>>>(probs, qkv, ctx);
  out_gemm<<<dim3(E_DIM / BN, M_TOT / BM), 256, 0, stream>>>(ctx, WoB, bo, out);
}
